// AttentionLayer_45964740002565
// MI455X (gfx1250) — matrix operand order repaired and checked by reading, not compiled, not run on hardware
//
#include <hip/hip_runtime.h>

// ---------- types ----------
typedef __attribute__((ext_vector_type(16))) __bf16        v16bf;
typedef __attribute__((ext_vector_type(8)))  float         v8f;
typedef __attribute__((ext_vector_type(4)))  unsigned int  u32x4;
typedef __attribute__((ext_vector_type(4)))  float         f32x4;

union BFPack { u32x4 u[2]; v16bf v; };

#define BATCH 8
#define SEQ   2048
#define DMODEL 512
#define UDIM  128

__device__ __forceinline__ unsigned short f2bf(float f) {
    unsigned int u = __float_as_uint(f);
    u = u + 0x7FFFu + ((u >> 16) & 1u);          // round-to-nearest-even
    return (unsigned short)(u >> 16);
}
__device__ __forceinline__ unsigned int pack_bf16(float a, float b) {
    unsigned int ua = __float_as_uint(a), ub = __float_as_uint(b);
    ua = ua + 0x7FFFu + ((ua >> 16) & 1u);
    ub = ub + 0x7FFFu + ((ub >> 16) & 1u);
    return (ua >> 16) | (ub & 0xFFFF0000u);
}
__device__ __forceinline__ v8f vzero8() {
    v8f z = {0.f,0.f,0.f,0.f,0.f,0.f,0.f,0.f};
    return z;
}
__device__ __forceinline__ v8f wmma_bf16(const BFPack& a, const BFPack& b, v8f c) {
    return __builtin_amdgcn_wmma_f32_16x16x32_bf16(
        false, a.v, false, b.v, (short)0, c, false, false);
}
__device__ __forceinline__ BFPack ldB(const unsigned short* p) {
    BFPack b;
    const u32x4* q = (const u32x4*)p;
    b.u[0] = q[0]; b.u[1] = q[1];
    return b;
}
// A operand: seg [g*16, g*16+16) within a 32-elem k-step
__device__ __forceinline__ BFPack ldA(const unsigned short* p) {
    BFPack a;
    a.u[0] = *(const u32x4*)(p);
    a.u[1] = *(const u32x4*)(p + 8);
    return a;
}

// ---------- kernel 0: X f32 -> bf16 (row-major, same indexing) ----------
__global__ __launch_bounds__(256)
void conv_x(const float* __restrict__ X, unsigned short* __restrict__ Xb) {
    long i = ((long)blockIdx.x * 256 + threadIdx.x) * 8;   // 8 floats per thread
    f32x4 a = *(const f32x4*)(X + i);
    f32x4 b = *(const f32x4*)(X + i + 4);
    u32x4 o;
    o.x = pack_bf16(a.x, a.y); o.y = pack_bf16(a.z, a.w);
    o.z = pack_bf16(b.x, b.y); o.w = pack_bf16(b.z, b.w);
    *(u32x4*)(Xb + i) = o;
}

// ---------- kernel 1: weight transpose + f32->bf16 ----------
// WqT/WkT/WvT: [U][D] bf16 (Wq scaled by log2(e)/sqrt(U));  WoT: [D][U] bf16
__global__ __launch_bounds__(256)
void prep_weights(const float* __restrict__ Wq, const float* __restrict__ Wk,
                  const float* __restrict__ Wv, const float* __restrict__ Wo,
                  unsigned short* __restrict__ WqT, unsigned short* __restrict__ WkT,
                  unsigned short* __restrict__ WvT, unsigned short* __restrict__ WoT) {
    int idx = blockIdx.x * 256 + threadIdx.x;       // 0 .. 4*65536-1
    int which = idx >> 16;
    int e = idx & 65535;
    if (which < 3) {
        int u = e >> 9;                 // 0..127
        int d = e & 511;                // 0..511
        const float* W = (which == 0) ? Wq : (which == 1) ? Wk : Wv;
        unsigned short* WT = (which == 0) ? WqT : (which == 1) ? WkT : WvT;
        // fold 1/sqrt(U) * log2(e) into Wq so softmax can use native exp2
        float s = (which == 0) ? (0.08838834764831845f * 1.4426950408889634f) : 1.0f;
        WT[u * DMODEL + d] = f2bf(W[d * UDIM + u] * s);
    } else {
        int d = e >> 7;                 // 0..511
        int u = e & 127;                // 0..127
        WoT[d * UDIM + u] = f2bf(Wo[u * DMODEL + d]);
    }
}

// ---------- kernel 2: QKV projection, one matrix per block ----------
// depth-1 software pipeline with sched_barrier-pinned phases:
//   [issue loads k+1] | [WMMA chain k]
__global__ __launch_bounds__(32)
void proj_qkv(const unsigned short* __restrict__ Xb,
              const unsigned short* __restrict__ WqT, const unsigned short* __restrict__ WkT,
              const unsigned short* __restrict__ WvT,
              unsigned short* __restrict__ Q, unsigned short* __restrict__ K,
              unsigned short* __restrict__ VT) {
    const int lane = threadIdx.x;
    const int g = lane >> 4, ln = lane & 15;
    const int bid = blockIdx.x;              // 0..1023 (B*N/16 strips)
    const int mat = blockIdx.y;              // 0..2  (Q, K, V)
    const int b = bid >> 7;                  // batch
    const int t = bid & 127;                 // row tile in batch
    const long rowg = (long)bid * 16;        // global row among B*N

    const unsigned short* WT = (mat == 0) ? WqT : (mat == 1) ? WkT : WvT;
    const unsigned short* xrow = Xb + (rowg + ln) * DMODEL + g * 16;

    v8f acc[8];
#pragma unroll
    for (int uc = 0; uc < 8; ++uc) acc[uc] = vzero8();

    BFPack a0 = ldA(xrow);
    BFPack BtA[8];
#pragma unroll
    for (int uc = 0; uc < 8; ++uc)
        BtA[uc] = ldB(WT + (long)(uc * 16 + ln) * DMODEL + g * 16);

#pragma unroll
    for (int ks = 0; ks < 16; ++ks) {
        BFPack a1 = a0;
        BFPack BtB[8];
#pragma unroll
        for (int uc = 0; uc < 8; ++uc) BtB[uc] = BtA[uc];
        if (ks < 15) {
            a1 = ldA(xrow + (ks + 1) * 32);
#pragma unroll
            for (int uc = 0; uc < 8; ++uc)
                BtB[uc] = ldB(WT + (long)(uc * 16 + ln) * DMODEL + (ks + 1) * 32 + g * 16);
        }
        __builtin_amdgcn_sched_barrier(0);   // keep next-step loads above the WMMA chain
#pragma unroll
        for (int uc = 0; uc < 8; ++uc)
            acc[uc] = wmma_bf16(a0, BtA[uc], acc[uc]);
        __builtin_amdgcn_sched_barrier(0);
        a0 = a1;
#pragma unroll
        for (int uc = 0; uc < 8; ++uc) BtA[uc] = BtB[uc];
    }

    if (mat < 2) {
        unsigned short* O = (mat == 0) ? Q : K;
#pragma unroll
        for (int uc = 0; uc < 8; ++uc) {
            int u = uc * 16 + ln;
#pragma unroll
            for (int r = 0; r < 8; ++r) {
                long row = rowg + r + 8 * g;
                O[row * UDIM + u] = f2bf(acc[uc][r]);
            }
        }
    } else {
#pragma unroll
        for (int uc = 0; uc < 8; ++uc) {
            int u = uc * 16 + ln;
            // transposed V: rows r+8g are contiguous along N -> one b128 store
            u32x4 pv;
            pv.x = pack_bf16(acc[uc][0], acc[uc][1]); pv.y = pack_bf16(acc[uc][2], acc[uc][3]);
            pv.z = pack_bf16(acc[uc][4], acc[uc][5]); pv.w = pack_bf16(acc[uc][6], acc[uc][7]);
            *(u32x4*)(VT + ((long)(b * UDIM + u)) * SEQ + t * 16 + 8 * g) = pv;
        }
    }
}

// ---------- kernel 3: flash attention (4-wave split-K) + output proj + residual ----------
__global__ __launch_bounds__(128)
void attn_out(const unsigned short* __restrict__ Q, const unsigned short* __restrict__ K,
              const unsigned short* __restrict__ VT, const unsigned short* __restrict__ WoT,
              const float* __restrict__ X, const float* __restrict__ b_o,
              float* __restrict__ Out) {
    __shared__ float accS[4][16][128];           // 32 KiB: per-wave unnormalized ctx
    __shared__ float mS[4][16], lS[4][16], lTotS[16];
    __shared__ unsigned short PlAll[4][16 * 32]; // per-wave P bounce
    __shared__ unsigned short Cl[16 * 128];      // combined ctx (bf16, A layout source)

    const int tid  = threadIdx.x;
    const int w    = tid >> 5;                   // wave id 0..3
    const int lane = tid & 31;
    const int g = lane >> 4, ln = lane & 15;
    const int bid = blockIdx.x;                  // 0..1023
    const int b = bid >> 7;
    const long rowg = (long)bid * 16;            // global row among B*N
    unsigned short* Pl = &PlAll[w][0];

    // Q A operands (16 x 128), pre-scaled by log2(e)/sqrt(U)
    BFPack qa[4];
    const unsigned short* qrow = Q + (rowg + ln) * UDIM + g * 16;
#pragma unroll
    for (int ks = 0; ks < 4; ++ks) qa[ks] = ldA(qrow + ks * 32);

    float m_i[8], l_i[8];
    v8f acc[8];
#pragma unroll
    for (int r = 0; r < 8; ++r) { m_i[r] = -1.0e30f; l_i[r] = 0.f; }
#pragma unroll
    for (int c = 0; c < 8; ++c) acc[c] = vzero8();

    const unsigned short* Kbase = K + (long)b * SEQ * UDIM;
    const unsigned short* Vbase = VT + (long)b * UDIM * SEQ;

    // this wave covers key columns [w*512, (w+1)*512) as 16 chunks of 32
    const int colBeg = w * 512;

    // prologue: K tiles for the first chunk
    BFPack ktA[2][4];
#pragma unroll
    for (int ns = 0; ns < 2; ++ns)
#pragma unroll
        for (int ks = 0; ks < 4; ++ks)
            ktA[ns][ks] = ldB(Kbase + (long)(colBeg + ns * 16 + ln) * UDIM + ks * 32 + g * 16);

    for (int jj = 0; jj < 16; ++jj) {
        const int col0 = colBeg + jj * 32;

        // phase 1: issue V tiles for this chunk (consumed at the bottom)
        BFPack vt[8];
#pragma unroll
        for (int c = 0; c < 8; ++c)
            vt[c] = ldB(Vbase + (long)(c * 16 + ln) * SEQ + col0 + g * 16);
        __builtin_amdgcn_sched_barrier(0);

        // phase 2: scores from register-resident ktA (oldest loads -> cheap wait)
        v8f s0 = vzero8(), s1 = vzero8();
#pragma unroll
        for (int ks = 0; ks < 4; ++ks) {
            s0 = wmma_bf16(qa[ks], ktA[0][ks], s0);
            s1 = wmma_bf16(qa[ks], ktA[1][ks], s1);
        }
        __builtin_amdgcn_sched_barrier(0);

        // phase 3: issue next chunk's K tiles
        BFPack ktB[2][4];
#pragma unroll
        for (int ns = 0; ns < 2; ++ns)
#pragma unroll
            for (int ks = 0; ks < 4; ++ks) ktB[ns][ks] = ktA[ns][ks];
        if (jj < 15) {
#pragma unroll
            for (int ns = 0; ns < 2; ++ns)
#pragma unroll
                for (int ks = 0; ks < 4; ++ks)
                    ktB[ns][ks] = ldB(Kbase + (long)(col0 + 32 + ns * 16 + ln) * UDIM + ks * 32 + g * 16);
        }
        __builtin_amdgcn_sched_barrier(0);

        // phase 4: online softmax (base-2; scale folded into Wq)
#pragma unroll
        for (int r = 0; r < 8; ++r) {
            float mx = fmaxf(s0[r], s1[r]);
#pragma unroll
            for (int off = 1; off < 16; off <<= 1) mx = fmaxf(mx, __shfl_xor(mx, off, 32));
            float mnew = fmaxf(m_i[r], mx);
            float alpha = exp2f(m_i[r] - mnew);
            float p0 = exp2f(s0[r] - mnew);
            float p1 = exp2f(s1[r] - mnew);
            float ps = p0 + p1;
#pragma unroll
            for (int off = 1; off < 16; off <<= 1) ps += __shfl_xor(ps, off, 32);
            l_i[r] = l_i[r] * alpha + ps;
            m_i[r] = mnew;
#pragma unroll
            for (int c = 0; c < 8; ++c) acc[c][r] *= alpha;
            Pl[(r + 8 * g) * 32 + ln]      = f2bf(p0);
            Pl[(r + 8 * g) * 32 + 16 + ln] = f2bf(p1);
        }
        // wave-private LDS bounce: DS ops are in-order within a wave;
        // fences stop the compiler from reordering stores/loads
        __threadfence_block();
        BFPack pa;
        pa.u[0] = *(const u32x4*)(Pl + ln * 32 + g * 16);
        pa.u[1] = *(const u32x4*)(Pl + ln * 32 + g * 16 + 8);
        __threadfence_block();

        // phase 5: PV accumulate (vt loads have had the whole softmax to land)
#pragma unroll
        for (int c = 0; c < 8; ++c)
            acc[c] = wmma_bf16(pa, vt[c], acc[c]);

#pragma unroll
        for (int ns = 0; ns < 2; ++ns)
#pragma unroll
            for (int ks = 0; ks < 4; ++ks) ktA[ns][ks] = ktB[ns][ks];
    }

    // ---- exact flash combine across the 4 waves ----
    if (ln == 0) {
#pragma unroll
        for (int r = 0; r < 8; ++r) { mS[w][r + 8 * g] = m_i[r]; lS[w][r + 8 * g] = l_i[r]; }
    }
    __syncthreads();
#pragma unroll
    for (int r = 0; r < 8; ++r) {
        int row = r + 8 * g;
        float mt = fmaxf(fmaxf(mS[0][row], mS[1][row]), fmaxf(mS[2][row], mS[3][row]));
        float lt = lS[0][row] * exp2f(mS[0][row] - mt) + lS[1][row] * exp2f(mS[1][row] - mt)
                 + lS[2][row] * exp2f(mS[2][row] - mt) + lS[3][row] * exp2f(mS[3][row] - mt);
        float factor = exp2f(m_i[r] - mt);
        if (w == 0 && ln == 0) lTotS[row] = lt;
#pragma unroll
        for (int c = 0; c < 8; ++c)
            accS[w][row][c * 16 + ln] = acc[c][r] * factor;
    }
    __syncthreads();
    {   // 128 threads x 16 elements: sum 4 partials, normalize, emit bf16 ctx
        int m = tid >> 3;
        int u0 = (tid & 7) * 16;
        float invl = 1.0f / lTotS[m];
#pragma unroll
        for (int i = 0; i < 16; ++i) {
            int u = u0 + i;
            float s = accS[0][m][u] + accS[1][m][u] + accS[2][m][u] + accS[3][m][u];
            Cl[m * 128 + u] = f2bf(s * invl);
        }
    }
    __syncthreads();

    // ---- output projection: wave w handles column tiles [w*8, w*8+8) ----
    BFPack ca[4];
#pragma unroll
    for (int ks = 0; ks < 4; ++ks)
        ca[ks] = ldA((const unsigned short*)Cl + ln * 128 + ks * 32 + g * 16);
    for (int nt = w * 8; nt < w * 8 + 8; ++nt) {
        int n = nt * 16 + ln;
        BFPack wt[4];
#pragma unroll
        for (int ks = 0; ks < 4; ++ks)
            wt[ks] = ldB(WoT + (long)n * UDIM + ks * 32 + g * 16);
        float bias = b_o[n];
        float xr[8];
#pragma unroll
        for (int r = 0; r < 8; ++r)
            xr[r] = X[(rowg + r + 8 * g) * DMODEL + n];
        __builtin_amdgcn_sched_barrier(0);

        v8f o = vzero8();
#pragma unroll
        for (int ks = 0; ks < 4; ++ks)
            o = wmma_bf16(ca[ks], wt[ks], o);

#pragma unroll
        for (int r = 0; r < 8; ++r) {
            long idx = (rowg + r + 8 * g) * DMODEL + n;
            Out[idx] = o[r] + bias + xr[r];
        }
    }
}

// ---------- launcher ----------
extern "C" void kernel_launch(void* const* d_in, const int* in_sizes, int n_in,
                              void* d_out, int out_size, void* d_ws, size_t ws_size,
                              hipStream_t stream) {
    (void)in_sizes; (void)n_in; (void)out_size; (void)ws_size;
    const float* X   = (const float*)d_in[0];
    const float* Wq  = (const float*)d_in[1];
    const float* Wk  = (const float*)d_in[2];
    const float* Wv  = (const float*)d_in[3];
    const float* Wo  = (const float*)d_in[4];
    const float* b_o = (const float*)d_in[5];
    float* Out = (float*)d_out;

    char* w = (char*)d_ws;
    unsigned short* WqT = (unsigned short*)(w);                    // 128*512*2 = 128 KiB each
    unsigned short* WkT = (unsigned short*)(w + (1u << 17));
    unsigned short* WvT = (unsigned short*)(w + (2u << 17));
    unsigned short* WoT = (unsigned short*)(w + (3u << 17));
    unsigned short* Qs  = (unsigned short*)(w + (4u << 17));       // 16384*128*2 = 4 MiB each
    unsigned short* Ks  = (unsigned short*)(w + (4u << 17) + (1u << 22));
    unsigned short* VT  = (unsigned short*)(w + (4u << 17) + (2u << 22));
    unsigned short* Xb  = (unsigned short*)(w + (4u << 17) + (3u << 22)); // 16 MiB

    conv_x<<<dim3(4096), dim3(256), 0, stream>>>(X, Xb);
    prep_weights<<<dim3(1024), dim3(256), 0, stream>>>(Wq, Wk, Wv, Wo, WqT, WkT, WvT, WoT);
    proj_qkv<<<dim3(1024, 3), dim3(32), 0, stream>>>(Xb, WqT, WkT, WvT, Qs, Ks, VT);
    attn_out<<<dim3(1024), dim3(128), 0, stream>>>(Qs, Ks, VT, WoT, X, b_o, Out);
}
